// GNNSiamese_47837345743302
// MI455X (gfx1250) — matrix operand order repaired and checked by reading, not compiled, not run on hardware
//
#include <hip/hip_runtime.h>
#include <stdint.h>
#include <stddef.h>

// ---------------- problem constants (from reference) ----------------
#define NNODES   25600
#define NEDGES   204800
#define NHEADS   4
#define NGRAPH   256          // BATCH*NG
#define NODES_PER_G 100       // NNODES / NGRAPH
#define NG       32
#define NBATCH   8

// ---------------- vector types ----------------
typedef __bf16  v16bf __attribute__((ext_vector_type(16)));
typedef float   v8f   __attribute__((ext_vector_type(8)));
typedef unsigned int   u32x4 __attribute__((ext_vector_type(4)));
typedef unsigned short u16x4 __attribute__((ext_vector_type(4)));
typedef float   f32x4 __attribute__((ext_vector_type(4)));

// ---------------- helpers ----------------
__device__ __forceinline__ uint16_t f2bf(float f) {
    uint32_t u = __float_as_uint(f);
    uint32_t r = u + 0x7FFFu + ((u >> 16) & 1u);   // round-to-nearest-even
    return (uint16_t)(r >> 16);
}
// monotone float <-> uint encoding so integer atomicMax == float max
__device__ __forceinline__ unsigned f_enc(float f) {
    unsigned u = __float_as_uint(f);
    return (u & 0x80000000u) ? ~u : (u | 0x80000000u);
}
__device__ __forceinline__ float f_dec(unsigned u) {
    unsigned v = (u & 0x80000000u) ? (u & 0x7FFFFFFFu) : ~u;
    return __uint_as_float(v);
}
__device__ __forceinline__ float lrelu(float v) { return v > 0.f ? v : 0.2f * v; }

// async global -> LDS copy of one 16B chunk (gfx1250 ASYNCcnt path)
__device__ __forceinline__ void async_b128(unsigned lds_addr, const void* gaddr) {
    asm volatile("global_load_async_to_lds_b128 %0, %1, off"
                 :: "v"(lds_addr), "v"((uint64_t)gaddr) : "memory");
}
__device__ __forceinline__ void wait_async0() {
    asm volatile("s_wait_asynccnt 0x0" ::: "memory");
}

// ---------------- kernels ----------------

// f32 -> bf16 elementwise, 4-wide (n multiple of 4)
__global__ void k_cvt_bf16(const float* __restrict__ x, uint16_t* __restrict__ xb, int n) {
    int i = (blockIdx.x * blockDim.x + threadIdx.x) * 4;
    if (i >= n) return;
    f32x4 v = *(const f32x4*)(x + i);
    u16x4 o;
#pragma unroll
    for (int t = 0; t < 4; ++t) o[t] = f2bf(v[t]);
    *(u16x4*)(xb + i) = o;
}

// Pack W [K, Nout] f32 row-major into WMMA B-fragment order:
// Bp[kb][ntile][lane][i] = bf16(W[kb*32 + (lane>>4)*16 + i][ntile*16 + (lane&15)])
__global__ void k_pack_w(const float* __restrict__ W, uint16_t* __restrict__ Bp,
                         int K, int Nout, int Ntiles) {
    int idx = blockIdx.x * blockDim.x + threadIdx.x;
    int total = (K >> 5) * Ntiles * 512;
    if (idx >= total) return;
    int i    = idx & 15;
    int lane = (idx >> 4) & 31;
    int nt   = (idx >> 9) % Ntiles;
    int kb   = idx / (512 * Ntiles);
    int n = nt * 16 + (lane & 15);
    int k = kb * 32 + (lane >> 4) * 16 + i;
    float v = (n < Nout) ? W[(size_t)k * Nout + n] : 0.f;
    Bp[idx] = f2bf(v);
}

// C[M,ldc] (f32) = A[M,K] (bf16 row-major) x Bp (packed bf16).
// Block = 4 waves covering (mtile, 4 consecutive ntiles).
// A tile (16 x K, contiguous) and the block's B fragments are staged in LDS via
// global_load_async_to_lds_b128 (ASYNCcnt); the WMMA loop runs entirely from LDS.
__global__ void __launch_bounds__(128)
k_gemm_wmma(const uint16_t* __restrict__ A, const uint16_t* __restrict__ Bp,
            float* __restrict__ C, int K, int Ntiles, int ldc) {
    __shared__ __align__(16) uint16_t Atile[16 * 256];     // 8 KB  (K <= 256)
    __shared__ __align__(16) uint16_t Btile[8 * 4 * 512];  // 32 KB (nkb <= 8, 4 tiles)

    int lane = threadIdx.x & 31;
    int wave = threadIdx.x >> 5;
    int nt0  = blockIdx.x * 4;
    int nt   = nt0 + wave;
    int mt   = blockIdx.y;
    int nkb  = K >> 5;

    // ---- stage A tile: 16 rows x K cols bf16 == one contiguous 32*K-byte block
    {
        unsigned lbase = (unsigned)(uintptr_t)(&Atile[0]);
        const uint16_t* g0 = A + (size_t)(mt * 16) * K;
        int nch = 2 * K;                               // 16-byte chunks
        for (int c = threadIdx.x; c < nch; c += 128)
            async_b128(lbase + (unsigned)c * 16, g0 + c * 8);
    }
    // ---- stage B fragments for this block's 4 n-tiles: [kb][w] -> 1 KB each
    {
        unsigned lbase = (unsigned)(uintptr_t)(&Btile[0]);
        int nch = nkb * 4 * 64;                        // (kb, w, 64 chunks of 16B)
        for (int c = threadIdx.x; c < nch; c += 128) {
            int cc = c & 63;
            int w  = (c >> 6) & 3;
            int kb = c >> 8;
            int ntw = nt0 + w;
            if (ntw < Ntiles)
                async_b128(lbase + (unsigned)c * 16,
                           Bp + (((size_t)kb * Ntiles + ntw) * 512) + cc * 8);
        }
    }
    wait_async0();
    __syncthreads();

    if (nt < Ntiles) {                                 // wave-uniform; EXEC all-ones inside
        int half = lane >> 4;
        v8f acc = {};
        const uint16_t* arow = &Atile[(lane & 15) * K];
        for (int kb = 0; kb < nkb; ++kb) {
            union { v16bf v; u32x4 q[2]; } a, b;
            const uint16_t* pa = arow + kb * 32 + half * 16;
            a.q[0] = *(const u32x4*)(pa);
            a.q[1] = *(const u32x4*)(pa + 8);
            const uint16_t* pb = &Btile[((kb * 4 + wave) * 512) + lane * 16];
            b.q[0] = *(const u32x4*)(pb);
            b.q[1] = *(const u32x4*)(pb + 8);
            acc = __builtin_amdgcn_wmma_f32_16x16x32_bf16(
                false, a.v, false, b.v, (short)0, acc, false, false);
        }
        int col = nt * 16 + (lane & 15);
        int r0  = mt * 16 + half * 8;
        if (col < ldc) {
#pragma unroll
            for (int v = 0; v < 8; ++v)
                C[(size_t)(r0 + v) * ldc + col] = acc[v];
        }
    }
}

// per (node, head): a_src/a_dst dot-products; seed segment-max with self-loop value
__global__ void k_attn(const float* __restrict__ h, const float* __restrict__ att_src,
                       const float* __restrict__ att_dst, float* __restrict__ a_src,
                       float* __restrict__ a_dst, unsigned* __restrict__ emax_u, int C) {
    int idx = blockIdx.x * blockDim.x + threadIdx.x;
    if (idx >= NNODES * NHEADS) return;
    int n = idx >> 2, hh = idx & 3;
    const float* hp = h + (size_t)n * (NHEADS * C) + hh * C;
    const float* as = att_src + hh * C;
    const float* ad = att_dst + hh * C;
    float s = 0.f, d = 0.f;
    for (int c = 0; c < C; ++c) { float hv = hp[c]; s += hv * as[c]; d += hv * ad[c]; }
    a_src[idx] = s; a_dst[idx] = d;
    emax_u[idx] = f_enc(lrelu(s + d));        // self-loop term
}

__global__ void k_edge_max(const int* __restrict__ src, const int* __restrict__ dst,
                           const float* __restrict__ a_src, const float* __restrict__ a_dst,
                           unsigned* __restrict__ emax_u) {
    int idx = blockIdx.x * blockDim.x + threadIdx.x;
    if (idx >= NEDGES * NHEADS) return;
    int e = idx >> 2, hh = idx & 3;
    int s = src[e], d = dst[e];
    float v = lrelu(a_src[s * 4 + hh] + a_dst[d * 4 + hh]);
    atomicMax(&emax_u[d * 4 + hh], f_enc(v));
}

// denom := exp(self_e - max); also keep self numerator
__global__ void k_init_den(const float* __restrict__ a_src, const float* __restrict__ a_dst,
                           const unsigned* __restrict__ emax_u, float* __restrict__ den,
                           float* __restrict__ snum) {
    int idx = blockIdx.x * blockDim.x + threadIdx.x;
    if (idx >= NNODES * NHEADS) return;
    float m = f_dec(emax_u[idx]);
    float num = __expf(lrelu(a_src[idx] + a_dst[idx]) - m);
    den[idx] = num;
    snum[idx] = num;
}

__global__ void k_edge_exp(const int* __restrict__ src, const int* __restrict__ dst,
                           const float* __restrict__ a_src, const float* __restrict__ a_dst,
                           const unsigned* __restrict__ emax_u, float* __restrict__ w_edge,
                           float* __restrict__ den) {
    int idx = blockIdx.x * blockDim.x + threadIdx.x;
    if (idx >= NEDGES * NHEADS) return;
    int e = idx >> 2, hh = idx & 3;
    int s = src[e], d = dst[e];
    float v = lrelu(a_src[s * 4 + hh] + a_dst[d * 4 + hh]);
    float w = __expf(v - f_dec(emax_u[d * 4 + hh]));
    w_edge[idx] = w;
    atomicAdd(&den[d * 4 + hh], w);
}

// out := h * alpha_self  (self-loop contribution; also initializes accumulator)
__global__ void k_init_out(const float* __restrict__ h, const float* __restrict__ snum,
                           const float* __restrict__ den, float* __restrict__ out,
                           int HC, int C) {
    long long idx = (long long)blockIdx.x * blockDim.x + threadIdx.x;
    if (idx >= (long long)NNODES * HC) return;
    int n = (int)(idx / HC), hc = (int)(idx % HC);
    int hh = hc / C;
    out[idx] = h[idx] * snum[n * 4 + hh] / den[n * 4 + hh];
}

// scatter-add messages: out[dst] += h[src] * (w_edge / den[dst])
__global__ void k_edge_scatter(const int* __restrict__ src, const int* __restrict__ dst,
                               const float* __restrict__ h, const float* __restrict__ w_edge,
                               const float* __restrict__ den, float* __restrict__ out,
                               int HC, int C) {
    int chunks = HC >> 2;
    long long idx = (long long)blockIdx.x * blockDim.x + threadIdx.x;
    if (idx >= (long long)NEDGES * chunks) return;
    int e = (int)(idx / chunks), j = (int)(idx % chunks);
    int s = src[e], d = dst[e];
    f32x4 hv = *(const f32x4*)(h + (size_t)s * HC + j * 4);
    float* op = out + (size_t)d * HC + j * 4;
#pragma unroll
    for (int t = 0; t < 4; ++t) {
        int hh = (j * 4 + t) / C;
        float w = w_edge[e * 4 + hh] / den[d * 4 + hh];
        atomicAdd(op + t, hv[t] * w);
    }
}

// x = relu(out + bias); emit bf16 copy (next GEMM) and/or f32 copy (final head); 4-wide
__global__ void k_relu_bias(const float* __restrict__ out, const float* __restrict__ bias,
                            uint16_t* __restrict__ xb, float* __restrict__ xf, int HC) {
    long long idx = ((long long)blockIdx.x * blockDim.x + threadIdx.x) * 4;
    if (idx >= (long long)NNODES * HC) return;
    f32x4 v = *(const f32x4*)(out + idx);
    f32x4 b = *(const f32x4*)(bias + (int)(idx % HC));
#pragma unroll
    for (int t = 0; t < 4; ++t) { v[t] += b[t]; v[t] = v[t] > 0.f ? v[t] : 0.f; }
    if (xb) {
        u16x4 o;
#pragma unroll
        for (int t = 0; t < 4; ++t) o[t] = f2bf(v[t]);
        *(u16x4*)(xb + idx) = o;
    }
    if (xf) *(f32x4*)(xf + idx) = v;
}

// head-mean + graph-mean pool + transposed store: out[br*256 + (g%32)*8 + g/32]
__global__ void k_pool(const float* __restrict__ xf, float* __restrict__ outp, int branch) {
    int g = blockIdx.x;          // 256 graphs, 32 threads each
    int t = threadIdx.x;
    float sum = 0.f;
    for (int i = t; i < NODES_PER_G; i += 32) {
        f32x4 hv = *(const f32x4*)(xf + (size_t)(g * NODES_PER_G + i) * 4);
        sum += (hv[0] + hv[1] + hv[2] + hv[3]) * 0.25f;
    }
#pragma unroll
    for (int o = 16; o > 0; o >>= 1) sum += __shfl_down(sum, o, 32);
    if (t == 0)
        outp[branch * NGRAPH + (g & 31) * NBATCH + (g >> 5)] = sum / (float)NODES_PER_G;
}

// ---------------- workspace layout (bytes) ----------------
static constexpr size_t OFF_XB   = 0;                          // bf16 x     [N,256]
static constexpr size_t OFF_H    = OFF_XB   + 13107200;        // f32 h      [N,256]
static constexpr size_t OFF_OUT  = OFF_H    + 26214400;        // f32 out    [N,256]
static constexpr size_t OFF_WP   = OFF_OUT  + 26214400;        // bf16 Wpack [256,256]
static constexpr size_t OFF_ASRC = OFF_WP   + 131072;
static constexpr size_t OFF_ADST = OFF_ASRC + 409600;
static constexpr size_t OFF_EMAX = OFF_ADST + 409600;
static constexpr size_t OFF_DEN  = OFF_EMAX + 409600;
static constexpr size_t OFF_SNUM = OFF_DEN  + 409600;
static constexpr size_t OFF_WED  = OFF_SNUM + 409600;          // f32 [E,4]
static constexpr size_t OFF_XF3  = OFF_WED  + 3276800;         // f32 [N,4]

static inline int cdiv(long long a, int b) { return (int)((a + b - 1) / b); }

extern "C" void kernel_launch(void* const* d_in, const int* in_sizes, int n_in,
                              void* d_out, int out_size, void* d_ws, size_t ws_size,
                              hipStream_t stream) {
    (void)in_sizes; (void)n_in; (void)out_size; (void)ws_size;
    const float* xin[2] = { (const float*)d_in[0], (const float*)d_in[3] };
    const int*   ein[2] = { (const int*)d_in[1],   (const int*)d_in[4]   };

    char* ws = (char*)d_ws;
    uint16_t* XB   = (uint16_t*)(ws + OFF_XB);
    float*    H    = (float*)   (ws + OFF_H);
    float*    OUT  = (float*)   (ws + OFF_OUT);
    uint16_t* WP   = (uint16_t*)(ws + OFF_WP);
    float*    ASRC = (float*)   (ws + OFF_ASRC);
    float*    ADST = (float*)   (ws + OFF_ADST);
    unsigned* EMAX = (unsigned*)(ws + OFF_EMAX);
    float*    DEN  = (float*)   (ws + OFF_DEN);
    float*    SNUM = (float*)   (ws + OFF_SNUM);
    float*    WED  = (float*)   (ws + OFF_WED);
    float*    XF3  = (float*)   (ws + OFF_XF3);
    float*    dout = (float*)d_out;

    const int dims[3][2] = { {128, 64}, {256, 64}, {256, 1} };  // (K, C)

    for (int br = 0; br < 2; ++br) {
        // convert input features to bf16 for the first GEMM
        int n0 = NNODES * 128;
        k_cvt_bf16<<<cdiv(n0 / 4, 256), 256, 0, stream>>>(xin[br], XB, n0);
        const int* srcp = ein[br];
        const int* dstp = ein[br] + NEDGES;

        for (int l = 0; l < 3; ++l) {
            int K = dims[l][0], C = dims[l][1];
            int HC = NHEADS * C;
            int Ntiles = (HC + 15) >> 4;
            const float* W     = (const float*)d_in[6 + 4 * l + 0];
            const float* att_s = (const float*)d_in[6 + 4 * l + 1];
            const float* att_d = (const float*)d_in[6 + 4 * l + 2];
            const float* bias  = (const float*)d_in[6 + 4 * l + 3];

            int wtot = (K >> 5) * Ntiles * 512;
            k_pack_w<<<cdiv(wtot, 256), 256, 0, stream>>>(W, WP, K, HC, Ntiles);

            dim3 gg(cdiv(Ntiles, 4), NNODES / 16);
            k_gemm_wmma<<<gg, 128, 0, stream>>>(XB, WP, H, K, Ntiles, HC);

            int nh = NNODES * NHEADS;
            k_attn<<<cdiv(nh, 256), 256, 0, stream>>>(H, att_s, att_d, ASRC, ADST, EMAX, C);
            int eh = NEDGES * NHEADS;
            k_edge_max<<<cdiv(eh, 256), 256, 0, stream>>>(srcp, dstp, ASRC, ADST, EMAX);
            k_init_den<<<cdiv(nh, 256), 256, 0, stream>>>(ASRC, ADST, EMAX, DEN, SNUM);
            k_edge_exp<<<cdiv(eh, 256), 256, 0, stream>>>(srcp, dstp, ASRC, ADST, EMAX, WED, DEN);

            long long nhc = (long long)NNODES * HC;
            k_init_out<<<cdiv(nhc, 256), 256, 0, stream>>>(H, SNUM, DEN, OUT, HC, C);
            long long esc = (long long)NEDGES * (HC >> 2);
            k_edge_scatter<<<cdiv(esc, 256), 256, 0, stream>>>(srcp, dstp, H, WED, DEN, OUT, HC, C);

            bool last = (l == 2);
            k_relu_bias<<<cdiv(nhc / 4, 256), 256, 0, stream>>>(
                OUT, bias, last ? nullptr : XB, last ? XF3 : nullptr, HC);
        }
        k_pool<<<NGRAPH, 32, 0, stream>>>(XF3, dout, br);
    }
}
